// encoder_1597727834447
// MI455X (gfx1250) — matrix operand order repaired and checked by reading, compiled, not run on hardware
//
#include <hip/hip_runtime.h>

typedef __attribute__((ext_vector_type(16))) _Float16 v16h;
typedef __attribute__((ext_vector_type(8)))  float    v8f;
typedef __attribute__((ext_vector_type(4)))  int      v4i;
typedef __attribute__((address_space(1))) v4i gv4i;   // global 16B chunk
typedef __attribute__((address_space(3))) v4i lv4i;   // LDS 16B chunk

#if __has_builtin(__builtin_amdgcn_global_load_async_to_lds_b128)
#define HAVE_ASYNC 1
#endif

#define BATCH 512
#define TT    127          // T-1
#define NN    256          // N
#define HH    256          // H
#define G4    1024         // 4H
#define BM    16           // batch rows per slice
#define NSLICE (BATCH/BM)  // 32 slices, x4 gate-WGs = 128 WGs
#define WPACK_HALVES (64*16*512)   // 64 col-tiles x 16 k-tiles x (32 lanes x 16 halves)
#define GATE_HALVES  (16*16*512)   // one gate's 256 columns: 256 KB in LDS

// workspace layout (bytes)
#define WS_ATTN  0
#define WS_WPACK (WS_ATTN + BATCH*NN*4)            // f16 packed [Wih;Whh]^T tiles
#define WS_BIAS  (WS_WPACK + WPACK_HALVES*2)       // b_ih + b_hh
#define WS_GS    (WS_BIAS + G4*4)                  // gate exchange scratch (B x 4H f32)
#define WS_CNT   (WS_GS + BATCH*G4*4)              // 32 barrier counters

// LDS: weights 256KB + A 16KB + bias 1KB + x-stage 16KB = 295936 B < 320 KB
#define REC_SMEM ((GATE_HALVES + 16*32*16)*2 + (NN + BM*NN)*4)

__device__ __forceinline__ float sigf(float v) { return 1.0f / (1.0f + __expf(-v)); }

// -------- pack [W_ih ; W_hh]^T into WMMA B-fragment layout (f16), fuse biases, zero counters
__global__ void pack_kernel(const float* __restrict__ Wih, const float* __restrict__ Whh,
                            const float* __restrict__ bih, const float* __restrict__ bhh,
                            _Float16* __restrict__ wpack, float* __restrict__ bias,
                            unsigned* __restrict__ counters) {
  int idx = blockIdx.x * blockDim.x + threadIdx.x;
  if (idx < WPACK_HALVES) {
    // flat = ((ct*16 + kt)*32 + lane)*16 + e
    int e  = idx & 15;
    int L  = (idx >> 4) & 31;
    int kt = (idx >> 9) & 15;
    int ct = idx >> 13;
    int j  = ct * 16 + (L & 15);                 // gate-matrix row (output column)
    int k  = kt * 32 + ((e >> 3) << 4) + ((L >> 4) << 3) + (e & 7);      // B-matrix K per ISA layout
    float v = (k < NN) ? Wih[j * NN + k] : Whh[j * HH + (k - NN)];
    wpack[idx] = (_Float16)v;
  }
  if (idx < G4)     bias[idx] = bih[idx] + bhh[idx];
  if (idx < NSLICE) counters[idx] = 0u;
}

// -------- attn = softmax_n( sum_t x[b,t,n]*w_x[t] + b_attn )   (shift-invariant => time-constant)
__global__ void attn_kernel(const float* __restrict__ x, const float* __restrict__ Wattn,
                            const float* __restrict__ battn, float* __restrict__ attn) {
  __shared__ float wx[TT];
  __shared__ float red[NN];
  int b = blockIdx.x, n = threadIdx.x;
  if (n < TT) wx[n] = Wattn[2 * HH + n];
  __syncthreads();
  const float* xb = x + (size_t)b * TT * NN + n;
  float s = battn[0];
  #pragma unroll 4
  for (int t = 0; t < TT; ++t) s += xb[(size_t)t * NN] * wx[t];
  red[n] = s; __syncthreads();
  for (int off = 128; off > 0; off >>= 1) { if (n < off) red[n] = fmaxf(red[n], red[n + off]); __syncthreads(); }
  float mx = red[0]; __syncthreads();
  float e = __expf(s - mx);
  red[n] = e; __syncthreads();
  for (int off = 128; off > 0; off >>= 1) { if (n < off) red[n] += red[n + off]; __syncthreads(); }
  attn[(size_t)b * NN + n] = e / red[0];
}

// -------- async-prefetch one x_t slice (BM x NN f32 = 16 KB) into LDS staging
__device__ __forceinline__ void load_xt(float* stage, const float* x, int b0, int t, int tid) {
  #pragma unroll
  for (int j = 0; j < 8; ++j) {
    int idx = tid + (j << 7);          // 16-byte chunk id, 1024 chunks total
    int m = idx >> 6;
    int n = (idx & 63) << 2;
    const float* src = &x[((size_t)(b0 + m) * TT + t) * NN + n];
    float* dst = stage + (idx << 2);
#ifdef HAVE_ASYNC
    __builtin_amdgcn_global_load_async_to_lds_b128((gv4i*)src, (lv4i*)dst, 0, 0);
#else
    *(v4i*)dst = *(const v4i*)src;
#endif
  }
}

__device__ __forceinline__ void wait_async() {
#ifdef HAVE_ASYNC
#if __has_builtin(__builtin_amdgcn_s_wait_asynccnt)
  __builtin_amdgcn_s_wait_asynccnt(0);
#else
  asm volatile("s_wait_asynccnt 0x0" ::: "memory");
#endif
#endif
}

// -------- inter-WG (4 gate-WGs per slice) generation barrier through L2
__device__ __forceinline__ void group_barrier(unsigned* cnt, unsigned target) {
  __threadfence();
  __syncthreads();
  if (threadIdx.x == 0) {
    atomicAdd(cnt, 1u);
    __builtin_amdgcn_s_cluster_barrier();   // s_barrier_signal/-wait -3; NOP when not in a cluster
    while (__hip_atomic_load(cnt, __ATOMIC_ACQUIRE, __HIP_MEMORY_SCOPE_AGENT) < target) {
      __builtin_amdgcn_s_sleep(1);
    }
  }
  __syncthreads();
  __threadfence();
}

// -------- recurrent LSTM: blockIdx.x = gate (i/f/g/o), blockIdx.y = batch slice
__global__ void __launch_bounds__(128, 1)
rec_kernel(const float* __restrict__ x, const _Float16* __restrict__ wpack,
           const float* __restrict__ attn, const float* __restrict__ bias,
           float* gs, unsigned* counters,
           float* __restrict__ outw, float* __restrict__ oute) {
  extern __shared__ char smem[];
  _Float16* lw    = (_Float16*)smem;            // 131072 halves: gate's weight slice (B-frag layout)
  _Float16* la    = lw + GATE_HALVES;           // 8192 halves: A = [weighted_t | h] (A-frag layout)
  float*    lbias = (float*)(la + 16 * 32 * 16);
  float*    stage = lbias + NN;                 // x_t staging (async target)

  const int gate  = blockIdx.x;
  const int slice = blockIdx.y;
  const int b0    = slice * BM;
  const int tid   = threadIdx.x;
  const int wv    = tid >> 5;       // wave 0..3
  const int L     = tid & 31;       // lane

  { // weights -> LDS once (resident for all 127 steps)
    const uint4* src = (const uint4*)(wpack + (size_t)gate * GATE_HALVES);
    uint4* dst = (uint4*)lw;
    for (int i = tid; i < GATE_HALVES / 8; i += 128) dst[i] = src[i];
  }
  for (int i = tid; i < NN;   i += 128) lbias[i] = bias[gate * NN + i];
  for (int i = tid; i < 4096; i += 128) la[4096 + i] = (_Float16)0.0f;  // h region (kt 8..15) = 0

  float attnreg[32], c[32];
  #pragma unroll
  for (int i = 0; i < 32; ++i) {
    attnreg[i] = attn[(size_t)b0 * NN + tid + (i << 7)];
    c[i] = 0.0f;
  }

  unsigned* cnt = counters + slice;
  float* gsl = gs + (size_t)slice * (4 * BM * NN);
  volatile const float* gslv = gsl;
  unsigned gen = 0;

  load_xt(stage, x, b0, 0, tid);     // prefetch x_0
  __syncthreads();

  for (int t = 0; t < TT; ++t) {
    wait_async();
    __syncthreads();                 // stage[t] ready for all waves
    // weighted_t = attn * x_t  -> LDS A (kt 0..7, ISA A-frag layout) + output
    #pragma unroll 4
    for (int i = 0; i < 32; ++i) {
      int f = tid + (i << 7);
      int m = f >> 8, n = f & 255;
      float w = attnreg[i] * stage[f];
      int kt = n >> 5, kk = n & 31;
      int lane2 = m + (((kk >> 3) & 1) << 4);
      int e2    = (kk & 7) | (((kk >> 4) & 1) << 3);
      la[((kt << 5) + lane2) * 16 + e2] = (_Float16)w;
      if (gate == 0) outw[((size_t)(b0 + m) * TT + t) * NN + n] = w;
    }
    __syncthreads();                 // la ready; stage reads done
    if (t + 1 < TT) load_xt(stage, x, b0, t + 1, tid);   // overlap with GEMM + barriers

    // gates(BM x 256) += A(BM x 512) * Wgate^T, all operands in LDS
    v8f acc[4] = {};
    #pragma unroll
    for (int kt = 0; kt < 16; ++kt) {
      v16h a = *(const v16h*)(la + ((kt << 5) + L) * 16);
      #pragma unroll
      for (int nt = 0; nt < 4; ++nt) {
        int lct = (wv << 2) + nt;
        v16h bf = *(const v16h*)(lw + ((size_t)((lct << 4) + kt) << 9) + (size_t)L * 16);
        acc[nt] = __builtin_amdgcn_wmma_f32_16x16x32_f16(false, a, false, bf,
                                                         (short)0, acc[nt], false, false);
      }
    }
    // + bias, publish gate slice to L2 scratch
    #pragma unroll
    for (int nt = 0; nt < 4; ++nt) {
      int jj = ((wv << 2) + nt) * 16 + (L & 15);
      float bv = lbias[jj];
      int mrow0 = (L >> 4) << 3;
      #pragma unroll
      for (int d = 0; d < 8; ++d)
        gsl[(size_t)gate * (BM * NN) + (size_t)(d + mrow0) * NN + jj] = acc[nt][d] + bv;
    }
    asm volatile("s_wait_storecnt 0x0" ::: "memory");
    group_barrier(cnt, 4 * (++gen));

    // elementwise LSTM update (each gate-WG redundantly, keeps h/c local)
    #pragma unroll 4
    for (int i = 0; i < 32; ++i) {
      int f = tid + (i << 7);
      float ig = gslv[f];
      float fg = gslv[1 * BM * NN + f];
      float gg = gslv[2 * BM * NN + f];
      float og = gslv[3 * BM * NN + f];
      float cn = sigf(fg) * c[i] + sigf(ig) * tanhf(gg);
      c[i] = cn;
      float h = sigf(og) * tanhf(cn);
      int m = f >> 8, hh = f & 255;
      int kt = 8 + (hh >> 5), kk = hh & 31;
      int lane2 = m + (((kk >> 3) & 1) << 4);
      int e2    = (kk & 7) | (((kk >> 4) & 1) << 3);
      la[((kt << 5) + lane2) * 16 + e2] = (_Float16)h;   // h -> A region for next step
      if (gate == 0) oute[((size_t)(b0 + m) * TT + t) * NN + hh] = h;
    }
    group_barrier(cnt, 4 * (++gen));   // readers done before next step's writes
  }
}

extern "C" void kernel_launch(void* const* d_in, const int* in_sizes, int n_in,
                              void* d_out, int out_size, void* d_ws, size_t ws_size,
                              hipStream_t stream) {
  const float* x     = (const float*)d_in[0];
  const float* Wattn = (const float*)d_in[1];
  const float* battn = (const float*)d_in[2];
  const float* Wih   = (const float*)d_in[3];
  const float* Whh   = (const float*)d_in[4];
  const float* bih   = (const float*)d_in[5];
  const float* bhh   = (const float*)d_in[6];
  float* out = (float*)d_out;

  char* ws = (char*)d_ws;
  float*    attn  = (float*)(ws + WS_ATTN);
  _Float16* wpack = (_Float16*)(ws + WS_WPACK);
  float*    bias  = (float*)(ws + WS_BIAS);
  float*    gs    = (float*)(ws + WS_GS);
  unsigned* cnt   = (unsigned*)(ws + WS_CNT);

  pack_kernel<<<(WPACK_HALVES + 255) / 256, 256, 0, stream>>>(Wih, Whh, bih, bhh, wpack, bias, cnt);
  attn_kernel<<<BATCH, NN, 0, stream>>>(x, Wattn, battn, attn);

  (void)hipFuncSetAttribute(reinterpret_cast<const void*>(rec_kernel),
                            hipFuncAttributeMaxDynamicSharedMemorySize, REC_SMEM);
  dim3 grid(4, NSLICE);
  rec_kernel<<<grid, 128, REC_SMEM, stream>>>(x, wpack, attn, bias, gs, cnt,
                                              out, out + (size_t)BATCH * TT * NN);
}
